// Gating_23356032156213
// MI455X (gfx1250) — matrix operand order repaired and checked by reading, not compiled, not run on hardware
//
#include <hip/hip_runtime.h>

// MI455X / gfx1250 fused gating kernel (fp16-fragment revision).
// Math: q,k1,k2,v1,v2 = proj(Q/K/V) [N,64]; 2-way softmax gate -> head [N,64];
//       out = head @ W_eff.T + lh_b, W_eff[j,d] = sum_h lh_w[j, h*64+d]  (tile-collapse).
// Single pass over Q/K/V (402 MB fp32 in, 134 MB fp32 out) => HBM-bound ~23us @ 23.3 TB/s.
// GEMMs via v_wmma_f32_16x16x32_f16 (11-bit mantissa vs bf16's 8: better accuracy,
// identical throughput); weights pre-swizzled to per-lane B-fragment order.

typedef __attribute__((ext_vector_type(16))) _Float16 v16h;
typedef __attribute__((ext_vector_type(8)))  _Float16 v8h;
typedef __attribute__((ext_vector_type(8)))  float    v8f;
typedef __attribute__((ext_vector_type(4)))  float    v4f;

#define IN_F   1024
#define D_HEAD 64

// ---------------------------------------------------------------------------
// Prep 1: pack the five [64,1024] projection weights into f16 WMMA B-fragments.
// B = W.T (K x N). Fragment (p, nt, ks): 32 lanes x 16 f16 (32B/lane).
// Lane L: n = nt*16 + (L&15), khalf = L>>4; elements e=0..15 hold
//   B[k0 + khalf*16 + e, n] = W[n, k0 + khalf*16 + e]  (contiguous in W's row).
// ---------------------------------------------------------------------------
__global__ void pack_proj_kernel(const float* __restrict__ w0, const float* __restrict__ w1,
                                 const float* __restrict__ w2, const float* __restrict__ w3,
                                 const float* __restrict__ w4, _Float16* __restrict__ projB) {
    int b    = blockIdx.x;        // 0..639 = 5 proj * 4 ntile * 32 kstep
    int p    = b >> 7;
    int rem  = b & 127;
    int nt   = rem >> 5;
    int ks   = rem & 31;
    int lane = threadIdx.x;       // 0..31
    const float* W = (p == 0) ? w0 : (p == 1) ? w1 : (p == 2) ? w2 : (p == 3) ? w3 : w4;
    int n  = nt * 16 + (lane & 15);
    int kh = lane >> 4;
    const float* src = W + (size_t)n * IN_F + ks * 32 + kh * 16;
    _Float16* dst = projB + ((size_t)((p * 4 + nt) * 32 + ks)) * 512 + lane * 16;
#pragma unroll
    for (int e = 0; e < 16; ++e) dst[e] = (_Float16)src[e];
}

// ---------------------------------------------------------------------------
// Prep 2: W_eff[j,d] = sum_h lh_w[j, h*64+d]; pack B2 = W_eff.T (64 x 1024)
// into f16 fragments (nt2 = 0..63 over the 1024 output cols, ks = 0..1 over K=64).
// ---------------------------------------------------------------------------
__global__ void pack_eff_kernel(const float* __restrict__ lhw, _Float16* __restrict__ effB) {
    int b    = blockIdx.x;        // 0..127
    int nt2  = b >> 1;
    int ks   = b & 1;
    int lane = threadIdx.x;
    int n    = nt2 * 16 + (lane & 15);   // output column j
    int kh   = lane >> 4;
    int kbase = ks * 32 + kh * 16;
    _Float16* dst = effB + ((size_t)(nt2 * 2 + ks)) * 512 + lane * 16;
#pragma unroll
    for (int e = 0; e < 16; ++e) {
        int k = kbase + e;               // d index 0..63
        float s = 0.f;
#pragma unroll
        for (int h = 0; h < 16; ++h) s += lhw[(size_t)n * IN_F + h * 64 + k];
        dst[e] = (_Float16)s;
    }
}

// A-fragment loader for 16-bit A 16x32 (ISA 7.12.2): lane L (m=L&15, h=L>>4)
// holds elements e=0..15 -> K = k0 + h*16 + e (contiguous, matches packed B).
__device__ __forceinline__ v16h load_a32(const float* __restrict__ row, int k0, int h) {
    const v4f* p0 = (const v4f*)(row + k0 + h * 16);
    v4f x0 = p0[0], x1 = p0[1];
    const v4f* p1 = (const v4f*)(row + k0 + h * 16 + 8);
    v4f y0 = p1[0], y1 = p1[1];
    v16h a;
#pragma unroll
    for (int i = 0; i < 4; ++i) {
        a[i]      = (_Float16)x0[i];
        a[4 + i]  = (_Float16)x1[i];
        a[8 + i]  = (_Float16)y0[i];
        a[12 + i] = (_Float16)y1[i];
    }
    return a;
}

// ---------------------------------------------------------------------------
// Main fused kernel: 1 wave == 16 rows. 256 blocks x 256 threads = 2048 waves
// = 32768/16 tiles exactly (EXEC always all-ones, as WMMA requires).
// ---------------------------------------------------------------------------
__global__ __launch_bounds__(256)
void gating_main_kernel(const float* __restrict__ Q, const float* __restrict__ K,
                        const float* __restrict__ V,
                        const float* __restrict__ lq_b,  const float* __restrict__ lk1_b,
                        const float* __restrict__ lk2_b, const float* __restrict__ lv1_b,
                        const float* __restrict__ lv2_b, const float* __restrict__ lh_b,
                        const _Float16* __restrict__ projB, const _Float16* __restrict__ effB,
                        float* __restrict__ out) {
    __shared__ _Float16 headLds[8 * 16 * 64];   // 2KB per wave, 16KB per WG

    int lane = threadIdx.x & 31;
    int wave = threadIdx.x >> 5;
    int tile = blockIdx.x * 8 + wave;
    int m0   = tile * 16;
    int mL   = lane & 15;
    int h    = lane >> 4;

    const float* qrow = Q + (size_t)(m0 + mL) * IN_F;
    const float* krow = K + (size_t)(m0 + mL) * IN_F;
    const float* vrow = V + (size_t)(m0 + mL) * IN_F;

    // Stage 1: five projections, K=1024 -> 32 WMMA k-steps, 4 N-tiles each.
    v8f acc[5][4];
#pragma unroll
    for (int p = 0; p < 5; ++p)
#pragma unroll
        for (int nt = 0; nt < 4; ++nt) acc[p][nt] = (v8f){0.f, 0.f, 0.f, 0.f, 0.f, 0.f, 0.f, 0.f};

    for (int ks = 0; ks < 32; ++ks) {
        int k0 = ks * 32;
        v16h aQ = load_a32(qrow, k0, h);
        v16h aK = load_a32(krow, k0, h);
        v16h aV = load_a32(vrow, k0, h);
        if (ks < 31)  // pull next k-step's weight line toward the WGP (global_prefetch_b8)
            __builtin_prefetch(projB + ((size_t)(ks + 1)) * 512 + lane * 16, 0, 1);
#pragma unroll
        for (int p = 0; p < 5; ++p) {
            v16h a = (p == 0) ? aQ : ((p == 1) || (p == 3)) ? aK : aV;
#pragma unroll
            for (int nt = 0; nt < 4; ++nt) {
                const v16h b = *(const v16h*)(projB + ((size_t)((p * 4 + nt) * 32 + ks)) * 512 + lane * 16);
                acc[p][nt] = __builtin_amdgcn_wmma_f32_16x16x32_f16(
                    false, a, false, b, (short)0, acc[p][nt], false, false);
            }
        }
    }

    // Bias add (C-layout: lane L, VGPR r -> element (m = r + 8*h, d = (L&15) + 16*nt)).
#pragma unroll
    for (int nt = 0; nt < 4; ++nt) {
        int d = mL + 16 * nt;
        float bq = lq_b[d], bk1 = lk1_b[d], bk2 = lk2_b[d], bv1 = lv1_b[d], bv2 = lv2_b[d];
#pragma unroll
        for (int r = 0; r < 8; ++r) {
            acc[0][nt][r] += bq;  acc[1][nt][r] += bk1; acc[2][nt][r] += bk2;
            acc[3][nt][r] += bv1; acc[4][nt][r] += bv2;
        }
    }

    // Gating: kv1 = <q,k1>, kv2 = <q,k2> per row; softmax over 2 == sigmoid.
    float head[4][8];
#pragma unroll
    for (int r = 0; r < 8; ++r) {
        float p1 = 0.f, p2 = 0.f;
#pragma unroll
        for (int nt = 0; nt < 4; ++nt) {
            p1 += acc[0][nt][r] * acc[1][nt][r];
            p2 += acc[0][nt][r] * acc[2][nt][r];
        }
#pragma unroll
        for (int s = 1; s < 16; s <<= 1) {   // reduce over the 16 lanes holding row m
            p1 += __shfl_xor(p1, s, 16);
            p2 += __shfl_xor(p2, s, 16);
        }
        float w1 = 1.f / (1.f + __expf(p2 - p1));
        float w2 = 1.f - w1;
#pragma unroll
        for (int nt = 0; nt < 4; ++nt)
            head[nt][r] = w1 * acc[3][nt][r] + w2 * acc[4][nt][r];
    }

    // Re-layout C-fragments -> A-fragments through per-wave LDS region.
    _Float16* hl = headLds + wave * (16 * 64);
#pragma unroll
    for (int nt = 0; nt < 4; ++nt)
#pragma unroll
        for (int r = 0; r < 8; ++r)
            hl[(r + 8 * h) * 64 + (mL + 16 * nt)] = (_Float16)head[nt][r];
    asm volatile("s_wait_dscnt 0x0" ::: "memory");   // same-wave DS RAW fence

    v16h aH[2];
#pragma unroll
    for (int ks2 = 0; ks2 < 2; ++ks2) {
        const _Float16* base = hl + mL * 64;
        v8h lo = *(const v8h*)(base + ks2 * 32 + h * 16);
        v8h hi = *(const v8h*)(base + ks2 * 32 + h * 16 + 8);
#pragma unroll
        for (int i = 0; i < 8; ++i) { aH[ks2][i] = lo[i]; aH[ks2][8 + i] = hi[i]; }
    }

    // Stage 2: out[16,1024] = head @ W_eff.T + lh_b; 64 N-tiles x 2 k-steps.
    for (int nt2 = 0; nt2 < 64; ++nt2) {
        v8f accO = (v8f){0.f, 0.f, 0.f, 0.f, 0.f, 0.f, 0.f, 0.f};
#pragma unroll
        for (int ks2 = 0; ks2 < 2; ++ks2) {
            const v16h b = *(const v16h*)(effB + ((size_t)(nt2 * 2 + ks2)) * 512 + lane * 16);
            accO = __builtin_amdgcn_wmma_f32_16x16x32_f16(
                false, aH[ks2], false, b, (short)0, accO, false, false);
        }
        int n = nt2 * 16 + mL;
        float bias = lh_b[n];
#pragma unroll
        for (int r = 0; r < 8; ++r)
            out[(size_t)(m0 + r + 8 * h) * IN_F + n] = accO[r] + bias;
    }
}

extern "C" void kernel_launch(void* const* d_in, const int* in_sizes, int n_in,
                              void* d_out, int out_size, void* d_ws, size_t ws_size,
                              hipStream_t stream) {
    const float* Q     = (const float*)d_in[0];
    const float* K     = (const float*)d_in[1];
    const float* V     = (const float*)d_in[2];
    const float* lq_w  = (const float*)d_in[3];
    const float* lq_b  = (const float*)d_in[4];
    const float* lk1_w = (const float*)d_in[5];
    const float* lk1_b = (const float*)d_in[6];
    const float* lk2_w = (const float*)d_in[7];
    const float* lk2_b = (const float*)d_in[8];
    const float* lv1_w = (const float*)d_in[9];
    const float* lv1_b = (const float*)d_in[10];
    const float* lv2_w = (const float*)d_in[11];
    const float* lv2_b = (const float*)d_in[12];
    const float* lh_w  = (const float*)d_in[13];
    const float* lh_b  = (const float*)d_in[14];
    float* out = (float*)d_out;

    _Float16* projB = (_Float16*)d_ws;                  // 5*128 frags * 512 f16 = 640 KB
    _Float16* effB  = projB + (size_t)5 * 128 * 512;    // 128 frags * 512 f16 = 128 KB

    pack_proj_kernel<<<640, 32, 0, stream>>>(lq_w, lk1_w, lk2_w, lv1_w, lv2_w, projB);
    pack_eff_kernel<<<128, 32, 0, stream>>>(lh_w, effB);
    gating_main_kernel<<<256, 256, 0, stream>>>(Q, K, V, lq_b, lk1_b, lk2_b,
                                                lv1_b, lv2_b, lh_b, projB, effB, out);
}
